// ScaledDotProductAttention_1443109012034
// MI455X (gfx1250) — compile-verified, matrix operand order checked by reading, not run on hardware
//
#include <hip/hip_runtime.h>
#include <hip/hip_bf16.h>

typedef __attribute__((ext_vector_type(16))) _Float16 v16h;
typedef __attribute__((ext_vector_type(8)))  float    v8f;

// Problem constants (reference: B=4, H=16, S=1024, D=64, temperature = 8)
#define BATCH 4
#define HEADS 16
#define SEQ   1024
#define DHEAD 64
#define INV_TEMP 0.125f

// LDS row stride in floats (1024 cols + 8 pad for bank spread, keeps 16B align)
#define LDST 1032
#define LDS_BYTES (64 * LDST * 4)   // 264192 B < 320 KB WGP limit

// Build a 16-half WMMA A/B fragment from 32 consecutive f32 "K" values of one
// row, honoring the CDNA5 16-bit fragment layout:
//   lanes 0-15 : halves 0-7 = K[0..7],   halves 8-15 = K[16..23]
//   lanes16-31 : halves 0-7 = K[8..15],  halves 8-15 = K[24..31]
__device__ __forceinline__ v16h frag_from_f32(const float* p, int lane) {
    const float* q = p + ((lane >> 4) << 3);
    v16h f;
#pragma unroll
    for (int i = 0; i < 8; ++i) {
        f[i]     = (_Float16)q[i];
        f[i + 8] = (_Float16)q[i + 16];
    }
    return f;
}

__device__ __forceinline__ v8f wmma_f16(v16h a, v16h b, v8f c) {
    return __builtin_amdgcn_wmma_f32_16x16x32_f16(
        /*neg_a=*/false, a, /*neg_b=*/false, b,
        /*c_mod=*/(short)0, c, /*reuse_a=*/false, /*reuse_b=*/false);
}

// One workgroup = 8 wave32 = 256 threads handles a 64-query tile of one (b,h).
__global__ void ScaledDotProductAttention_1443109012034_kernel(
    const float* __restrict__ q,
    const float* __restrict__ k,
    const float* __restrict__ v,
    const int*   __restrict__ mask,
    float* __restrict__ out,     // [B,H,S,D]
    float* __restrict__ attn)    // [B,H,S,S]
{
    extern __shared__ float lds[];   // [64][LDST] f32 scores -> probs

    const int lane = threadIdx.x & 31;
    const int wave = threadIdx.x >> 5;          // 0..7
    const int bh    = blockIdx.x >> 4;          // b*H + h  (16 qtiles per bh)
    const int qtile = blockIdx.x & 15;
    const int q0    = qtile * 64;
    const int b     = bh >> 4;                  // H == 16

    const size_t rowbase = (size_t)bh * SEQ;    // row index base into [BH, S, D]

    // ---------------- Phase 1: raw scores = Q (64x64) @ K^T (64x1024) -------
    {
        const int rt = wave >> 1;               // 16-row tile 0..3
        const float* qrow = q + (rowbase + q0 + rt * 16 + (lane & 15)) * DHEAD;
        const v16h aq0 = frag_from_f32(qrow,      lane);   // d = 0..31
        const v16h aq1 = frag_from_f32(qrow + 32, lane);   // d = 32..63

        for (int ct = (wave & 1); ct < SEQ / 16; ct += 2) {
            const float* krow = k + (rowbase + ct * 16 + (lane & 15)) * DHEAD;
            // Speculative prefetch of next K tile row (global_prefetch_b8)
            if (ct + 2 < SEQ / 16)
                __builtin_prefetch(krow + 2 * 16 * DHEAD, 0, 1);
            const v16h bk0 = frag_from_f32(krow,      lane);
            const v16h bk1 = frag_from_f32(krow + 32, lane);
            v8f c = {};
            c = wmma_f16(aq0, bk0, c);
            c = wmma_f16(aq1, bk1, c);
            // C layout: c[i] -> (M = (lane>=16?8:0)+i, N = lane&15)
            const int mb = rt * 16 + ((lane >> 4) << 3);
            const int n  = ct * 16 + (lane & 15);
#pragma unroll
            for (int i = 0; i < 8; ++i)
                lds[(size_t)(mb + i) * LDST + n] = c[i];
        }
    }
    __syncthreads();

    // ---------------- Phase 2: mask + scale + softmax, emit attn ------------
    {
#pragma unroll 1
        for (int j = 0; j < 8; ++j) {
            const int r = wave * 8 + j;         // local row 0..63
            float* row = lds + (size_t)r * LDST;
            const int* mrow = mask + (size_t)b * SEQ * SEQ
                                   + (size_t)(q0 + r) * SEQ;
            float mx = -3.4e38f;
            for (int i = lane; i < SEQ; i += 32) {
                float s = row[i];
                s = (mrow[i] == 0) ? -1e9f : s * INV_TEMP;
                row[i] = s;
                mx = fmaxf(mx, s);
            }
#pragma unroll
            for (int off = 16; off; off >>= 1)
                mx = fmaxf(mx, __shfl_xor(mx, off, 32));

            float sum = 0.0f;
            for (int i = lane; i < SEQ; i += 32) {
                float e = __expf(row[i] - mx);
                row[i] = e;
                sum += e;
            }
#pragma unroll
            for (int off = 16; off; off >>= 1)
                sum += __shfl_xor(sum, off, 32);

            const float inv = 1.0f / sum;
            float* arow = attn + (rowbase + (q0 + r)) * (size_t)SEQ;
            for (int i = lane; i < SEQ; i += 32) {
                float p = row[i] * inv;
                row[i]  = p;                          // kept in LDS for P @ V
                __builtin_nontemporal_store(p, arow + i);  // 268 MB stream: TH=NT
            }
        }
    }
    __syncthreads();

    // ---------------- Phase 3: O (64x64) = P (64x1024) @ V (1024x64) --------
    {
        const int rt = wave >> 1;               // 16-row tile 0..3
        const int dp = wave & 1;                // d-half: cols dp*32 .. dp*32+31
        const float* arow = lds + (size_t)(rt * 16 + (lane & 15)) * LDST;
        const float* vb   = v + rowbase * DHEAD;
        const int dc0 = dp * 32 + (lane & 15);
        const int dc1 = dc0 + 16;
        const int koff = (lane >> 4) << 3;

        v8f acc0 = {}, acc1 = {};
#pragma unroll 1
        for (int ns = 0; ns < SEQ / 32; ++ns) {
            const int n0 = ns * 32 + koff;
            // Prefetch next V chunk rows one iteration ahead
            if (ns + 1 < SEQ / 32)
                __builtin_prefetch(vb + (size_t)(n0 + 32) * DHEAD + dc0, 0, 1);
            const v16h a = frag_from_f32(arow + ns * 32, lane);
            v16h b0, b1;
#pragma unroll
            for (int i = 0; i < 8; ++i) {
                const float* r0 = vb + (size_t)(n0 + i) * DHEAD;
                const float* r1 = vb + (size_t)(n0 + 16 + i) * DHEAD;
                b0[i]     = (_Float16)r0[dc0];
                b1[i]     = (_Float16)r0[dc1];
                b0[i + 8] = (_Float16)r1[dc0];
                b1[i + 8] = (_Float16)r1[dc1];
            }
            acc0 = wmma_f16(a, b0, acc0);
            acc1 = wmma_f16(a, b1, acc1);
        }

        const int mb = rt * 16 + ((lane >> 4) << 3);
#pragma unroll
        for (int i = 0; i < 8; ++i) {
            float* orow = out + (rowbase + q0 + mb + i) * DHEAD;
            __builtin_nontemporal_store(acc0[i], orow + dp * 32 + (lane & 15));
            __builtin_nontemporal_store(acc1[i], orow + dp * 32 + 16 + (lane & 15));
        }
    }
}

extern "C" void kernel_launch(void* const* d_in, const int* in_sizes, int n_in,
                              void* d_out, int out_size, void* d_ws, size_t ws_size,
                              hipStream_t stream) {
    (void)in_sizes; (void)n_in; (void)d_ws; (void)ws_size; (void)out_size;
    const float* q    = (const float*)d_in[0];
    const float* k    = (const float*)d_in[1];
    const float* v    = (const float*)d_in[2];
    const int*   mask = (const int*)d_in[3];

    float* out  = (float*)d_out;                                    // [B,H,S,D]
    float* attn = out + (size_t)BATCH * HEADS * SEQ * DHEAD;        // [B,H,S,S]

    dim3 grid(BATCH * HEADS * (SEQ / 64));   // 1024 workgroups
    dim3 block(256);                         // 8 wave32
    ScaledDotProductAttention_1443109012034_kernel<<<grid, block, LDS_BYTES, stream>>>(
        q, k, v, mask, out, attn);
}
